// TaskMoE_Face_sixR_21723944583388
// MI455X (gfx1250) — matrix operand order repaired and checked by reading, not compiled, not run on hardware
//
#include <hip/hip_runtime.h>
#include <hip/hip_bf16.h>

// ---------------------------------------------------------------------------
// TaskMoE on MI455X (gfx1250): sparse top-2 dispatch, bf16-split WMMA GEMMs,
// double-buffered TDM (tensor_load_to_lds) weight tiles + async global->LDS
// activation tiles. Shapes: N=4096 tokens, D=H=1024, E=8 experts, top-2.
// ---------------------------------------------------------------------------

typedef __attribute__((ext_vector_type(16))) __bf16        v16bf;
typedef __attribute__((ext_vector_type(8)))  float         v8f;
typedef __attribute__((ext_vector_type(4)))  unsigned int  u32x4;
typedef __attribute__((ext_vector_type(8)))  int           i32x8;
typedef __attribute__((ext_vector_type(4)))  int           i32x4;

#define N_TOK 4096
#define DDIM  1024
#define HDIM  1024
#define NEXP  8
#define HPAD_ROWS (2 * N_TOK + 64)

#if defined(__has_builtin)
#if __has_builtin(__builtin_amdgcn_tensor_load_to_lds)
#define HAVE_TDM 1
#endif
#endif
#ifndef HAVE_TDM
#define HAVE_TDM 0
#endif

// ---- bf16 helpers ----------------------------------------------------------
__device__ __forceinline__ unsigned short f32_to_bf16(float v) {
  unsigned u = __float_as_uint(v);
  u += 0x7FFFu + ((u >> 16) & 1u);                 // round-to-nearest-even
  return (unsigned short)(u >> 16);
}
__device__ __forceinline__ void split_bf16(float v, unsigned short& hi,
                                           unsigned short& lo) {
  unsigned short h = f32_to_bf16(v);
  hi = h;
  lo = f32_to_bf16(v - __uint_as_float((unsigned)h << 16));
}

// ---- CDNA5 data movers -----------------------------------------------------
// per-lane 16B async copy global -> LDS (ASYNCcnt)
__device__ __forceinline__ void async_cp16(void* lds_dst, const void* gsrc) {
  unsigned l = (unsigned)(unsigned long long)lds_dst;   // LDS aperture: low 32b
  unsigned long long g = (unsigned long long)gsrc;
  asm volatile("global_load_async_to_lds_b128 %0, %1, off"
               :: "v"(l), "v"(g) : "memory");
}
__device__ __forceinline__ void wait_async0() {
  asm volatile("s_wait_asynccnt 0x0" ::: "memory");
}
__device__ __forceinline__ void wait_async2() {      // allow next tile in flight
  asm volatile("s_wait_asynccnt 0x2" ::: "memory");
}

// TDM: 2D tile load, 64 rows x 32 halfwords, row stride 1024 halfwords.
// D# per ISA 8.3/8.4: g0={count|flags, lds_addr, gaddr_lo, gaddr_hi|type=2},
// g1={mask/dsize, dim0, dim1, tiles, strides}, g2/g3 zero (2D tensor).
__device__ __forceinline__ void tdm_load_32x64(void* lds, const void* g) {
#if HAVE_TDM
  unsigned long long ga = (unsigned long long)g;
  u32x4 g0;
  g0[0] = 1u;                                       // count=1
  g0[1] = (unsigned)(unsigned long long)lds;        // lds_addr (bytes)
  g0[2] = (unsigned)ga;                             // global_addr[31:0]
  g0[3] = (unsigned)(ga >> 32) | (2u << 30);        // addr[56:32] | type=2
  i32x8 g1;
  g1[0] = 1 << 16;                                  // data_size=1 (2 bytes)
  g1[1] = (int)(1024u << 16);                       // tensor_dim0 = 1024
  g1[2] = 0;                                        // dim0_hi | dim1_lo
  g1[3] = 16 | (32 << 16);                          // dim1=1<<20 | tile_dim0=32
  g1[4] = 64;                                       // tile_dim1=64, tile_dim2=0
  g1[5] = 1024;                                     // tensor_dim0_stride
  g1[6] = 0;
  g1[7] = 0;
  i32x4 gz4 = {0, 0, 0, 0};
  i32x8 gz8 = {0, 0, 0, 0, 0, 0, 0, 0};
  __builtin_amdgcn_tensor_load_to_lds(g0, g1, gz4, gz4, gz8, 0);
#else
  (void)lds; (void)g;
#endif
}
__device__ __forceinline__ void wait_tensor0() {
#if HAVE_TDM
  __builtin_amdgcn_s_wait_tensorcnt(0);
#endif
}
__device__ __forceinline__ void wait_tensor2() {
#if HAVE_TDM
  __builtin_amdgcn_s_wait_tensorcnt(2);
#endif
}

// ---- WMMA fragment loaders (ISA 7.12.2 layouts) ----------------------------
__device__ __forceinline__ v16bf load_frag_a(const unsigned short* lA,
                                             int lane, int wm) {
  int row = wm + (lane & 15);
  int kb  = (lane >> 4) * 8;
  v16bf f;
  uint4* p = reinterpret_cast<uint4*>(&f);
  p[0] = *reinterpret_cast<const uint4*>(lA + row * 32 + kb);
  p[1] = *reinterpret_cast<const uint4*>(lA + row * 32 + 16 + kb);
  return f;
}
__device__ __forceinline__ v16bf load_frag_b(const unsigned short* lB,
                                             int lane, int wn) {
  int nn = wn + (lane & 15);
  int kb = (lane >> 4) * 8;
  v16bf f;
  uint4* p = reinterpret_cast<uint4*>(&f);
  p[0] = *reinterpret_cast<const uint4*>(lB + nn * 32 + kb);
  p[1] = *reinterpret_cast<const uint4*>(lB + nn * 32 + 16 + kb);
  return f;
}
__device__ __forceinline__ v8f wmma_split(v8f c, v16bf ah, v16bf al,
                                          v16bf bh, v16bf bl) {
  c = __builtin_amdgcn_wmma_f32_16x16x32_bf16(false, ah, false, bl, (short)0,
                                              c, false, false);
  c = __builtin_amdgcn_wmma_f32_16x16x32_bf16(false, al, false, bh, (short)0,
                                              c, false, false);
  c = __builtin_amdgcn_wmma_f32_16x16x32_bf16(false, ah, false, bh, (short)0,
                                              c, false, false);
  return c;
}

// ---- kernel 0: zero y ------------------------------------------------------
__global__ void moe_zero_y(float4* __restrict__ y) {
  size_t i = (size_t)blockIdx.x * blockDim.x + threadIdx.x;
  y[i] = make_float4(0.f, 0.f, 0.f, 0.f);
}

// ---- kernel 1: split x into bf16 hi/lo planes ------------------------------
__global__ void moe_prep_x(const float4* __restrict__ x4,
                           unsigned short* __restrict__ xh,
                           unsigned short* __restrict__ xl) {
  size_t i = (size_t)blockIdx.x * blockDim.x + threadIdx.x;
  float4 v = x4[i];
  unsigned short h0, l0, h1, l1, h2, l2, h3, l3;
  split_bf16(v.x, h0, l0); split_bf16(v.y, h1, l1);
  split_bf16(v.z, h2, l2); split_bf16(v.w, h3, l3);
  unsigned long long hp = (unsigned long long)h0 | ((unsigned long long)h1 << 16) |
                          ((unsigned long long)h2 << 32) | ((unsigned long long)h3 << 48);
  unsigned long long lp = (unsigned long long)l0 | ((unsigned long long)l1 << 16) |
                          ((unsigned long long)l2 << 32) | ((unsigned long long)l3 << 48);
  *reinterpret_cast<unsigned long long*>(xh + 4 * i) = hp;
  *reinterpret_cast<unsigned long long*>(xl + 4 * i) = lp;
}

// ---- kernel 2: split+transpose weights into [e][n][k] bf16 planes ----------
__global__ void moe_prep_w(const float* __restrict__ w1,
                           const float* __restrict__ w2,
                           unsigned short* __restrict__ w1h,
                           unsigned short* __restrict__ w1l,
                           unsigned short* __restrict__ w2h,
                           unsigned short* __restrict__ w2l) {
  __shared__ unsigned short th[64 * 64];
  __shared__ unsigned short tl[64 * 64];
  int z = blockIdx.z;                    // 0..15: [w1|w2] x expert
  int e = z & 7;
  const float* src = (z < 8 ? w1 : w2) + (size_t)e * DDIM * HDIM;
  unsigned short* dh = (z < 8 ? w1h : w2h) + (size_t)e * DDIM * HDIM;
  unsigned short* dl = (z < 8 ? w1l : w2l) + (size_t)e * DDIM * HDIM;
  int k0 = blockIdx.x * 64, n0 = blockIdx.y * 64;
  int t = threadIdx.x;
#pragma unroll
  for (int i = 0; i < 16; ++i) {
    int f = t + i * 256;
    int kk = f >> 6, nn = f & 63;
    float v = src[(size_t)(k0 + kk) * 1024 + n0 + nn];
    unsigned short hi, lo;
    split_bf16(v, hi, lo);
    th[nn * 64 + kk] = hi;
    tl[nn * 64 + kk] = lo;
  }
  __syncthreads();
#pragma unroll
  for (int i = 0; i < 16; ++i) {
    int f = t + i * 256;
    int nn = f >> 6, kk = f & 63;
    size_t o = (size_t)(n0 + nn) * 1024 + k0 + kk;
    dh[o] = th[nn * 64 + kk];
    dl[o] = tl[nn * 64 + kk];
  }
}

// ---- kernel 3: noisy top-2 gating (one wave32 per token) -------------------
__global__ void moe_gating(const float* __restrict__ x,
                           const float* __restrict__ wg,
                           const float* __restrict__ eps,
                           float* __restrict__ gates,
                           float* __restrict__ probs,
                           float* __restrict__ lse2) {
  int lane = threadIdx.x & 31;
  int wid  = threadIdx.x >> 5;
  int n = blockIdx.x * 8 + wid;
  float acc[16];
#pragma unroll
  for (int j = 0; j < 16; ++j) acc[j] = 0.f;
  const float* xr = x + (size_t)n * DDIM;
  for (int d0 = 0; d0 < DDIM; d0 += 32) {
    float xv = xr[d0 + lane];
#pragma unroll
    for (int j = 0; j < 16; ++j)
      acc[j] = fmaf(xv, wg[j * DDIM + d0 + lane], acc[j]);
  }
#pragma unroll
  for (int j = 0; j < 16; ++j) {
#pragma unroll
    for (int off = 16; off >= 1; off >>= 1)
      acc[j] += __shfl_xor(acc[j], off, 32);
  }
  if (lane == 0) {
    float logit[NEXP];
    float mx = -1e30f;
#pragma unroll
    for (int e = 0; e < NEXP; ++e) {
      float raw = acc[8 + e];
      float sp  = fmaxf(raw, 0.f) + log1pf(__expf(-fabsf(raw)));
      float lg  = acc[e] + eps[n * NEXP + e] * (sp + 0.01f);
      logit[e] = lg;
      mx = fmaxf(mx, lg);
    }
    float p[NEXP], s = 0.f;
#pragma unroll
    for (int e = 0; e < NEXP; ++e) { p[e] = __expf(logit[e] - mx); s += p[e]; }
    float inv = 1.f / s;
#pragma unroll
    for (int e = 0; e < NEXP; ++e) { p[e] *= inv; probs[n * NEXP + e] = p[e]; }
    int i1 = 0; float g1 = -1.f;
    for (int e = 0; e < NEXP; ++e) if (p[e] > g1) { g1 = p[e]; i1 = e; }
    int i2 = -1; float g2 = -1.f;
    for (int e = 0; e < NEXP; ++e)
      if (e != i1 && p[e] > g2) { g2 = p[e]; i2 = e; }
#pragma unroll
    for (int e = 0; e < NEXP; ++e) gates[n * NEXP + e] = 0.f;
    gates[n * NEXP + i1] = g1;
    gates[n * NEXP + i2] = g2;
    float lse = mx + __logf(s);
    lse2[n] = lse * lse;
  }
}

// ---- kernel 4: deterministic per-expert compaction + prefix ----------------
__global__ void moe_compact(const float* __restrict__ gates,
                            int* __restrict__ cnt, int* __restrict__ seg,
                            int* __restrict__ idxbuf) {
  __shared__ int scnt[NEXP];
  int lane = threadIdx.x & 31;
  int e    = threadIdx.x >> 5;
  int c = 0;
  for (int base = 0; base < N_TOK; base += 32) {
    int n = base + lane;
    bool act = gates[n * NEXP + e] > 0.f;
    unsigned mask = (unsigned)__ballot(act);
    if (act) idxbuf[e * N_TOK + c + __popc(mask & ((1u << lane) - 1u))] = n;
    c += __popc(mask);
  }
  if (lane == 0) { cnt[e] = c; scnt[e] = c; }
  __syncthreads();
  if (threadIdx.x == 0) {
    int o = 0;
    for (int i = 0; i < NEXP; ++i) { seg[i] = o; o += scnt[i]; }
  }
}

// ---- kernel 5: grouped GEMM1  h = relu(gather(x) @ w1[e]) ------------------
// 64x64 tile, 8 waves, bf16-split WMMA, double-buffered TDM+async staging.
__global__ void moe_ffn1(const unsigned short* __restrict__ xh,
                         const unsigned short* __restrict__ xl,
                         const unsigned short* __restrict__ wh,
                         const unsigned short* __restrict__ wl,
                         const int* __restrict__ cnt,
                         const int* __restrict__ seg,
                         const int* __restrict__ idxbuf,
                         unsigned short* __restrict__ hbh,
                         unsigned short* __restrict__ hbl) {
  __shared__ __align__(16) unsigned short lAh[2][2048], lAl[2][2048];
  __shared__ __align__(16) unsigned short lBh[2][2048], lBl[2][2048];
  __shared__ int stok[64];
  int e  = blockIdx.z;
  int mc = cnt[e];
  int m0 = blockIdx.x * 64;
  if (m0 >= mc) return;
  int n0 = blockIdx.y * 64;
  int so = seg[e];
  int t = threadIdx.x, lane = t & 31, wid = t >> 5;
  int wm = (wid & 3) * 16, wn = (wid >> 2) * 32;
  if (t < 64) {
    int a = m0 + t;
    stok[t] = idxbuf[e * N_TOK + (a < mc ? a : mc - 1)];  // clamp: garbage rows discarded
  }
  __syncthreads();
  int r = t >> 2, c8 = (t & 3) * 8;
  const unsigned short* ah = xh + (size_t)stok[r] * DDIM + c8;
  const unsigned short* al = xl + (size_t)stok[r] * DDIM + c8;
  const unsigned short* bh = wh + (size_t)e * DDIM * HDIM + (size_t)n0 * DDIM;
  const unsigned short* bl = wl + (size_t)e * DDIM * HDIM + (size_t)n0 * DDIM;
  v8f c0 = {0.f, 0.f, 0.f, 0.f, 0.f, 0.f, 0.f, 0.f};
  v8f c1 = c0;

  // prologue: tile 0 -> buffer 0
  if (wid == 0) {
    tdm_load_32x64(&lBh[0][0], bh);
    tdm_load_32x64(&lBl[0][0], bl);
  }
  async_cp16(&lAh[0][t * 8], ah);
  async_cp16(&lAl[0][t * 8], al);

  for (int it = 0; it < DDIM / 32; ++it) {
    int cur = it & 1, nxt = cur ^ 1;
    int k1 = (it + 1) * 32;
    if (k1 < DDIM) {                       // prefetch tile it+1 into other buf
      if (wid == 0) {
        tdm_load_32x64(&lBh[nxt][0], bh + k1);
        tdm_load_32x64(&lBl[nxt][0], bl + k1);
      }
      async_cp16(&lAh[nxt][t * 8], ah + k1);
      async_cp16(&lAl[nxt][t * 8], al + k1);
      wait_async2();                       // tile it done (in-order), it+1 in flight
      if (wid == 0) wait_tensor2();
    } else {
      wait_async0();
      if (wid == 0) wait_tensor0();
    }
    __syncthreads();
    v16bf fah = load_frag_a(&lAh[cur][0], lane, wm);
    v16bf fal = load_frag_a(&lAl[cur][0], lane, wm);
    v16bf bh0 = load_frag_b(&lBh[cur][0], lane, wn);
    v16bf bl0 = load_frag_b(&lBl[cur][0], lane, wn);
    v16bf bh1 = load_frag_b(&lBh[cur][0], lane, wn + 16);
    v16bf bl1 = load_frag_b(&lBl[cur][0], lane, wn + 16);
    c0 = wmma_split(c0, fah, fal, bh0, bl0);
    c1 = wmma_split(c1, fah, fal, bh1, bl1);
    __syncthreads();
  }

  int ml = (lane >> 4) * 8, nl = lane & 15;
#pragma unroll
  for (int v = 0; v < 8; ++v) {
    int a = m0 + wm + ml + v;
    if (a < mc) {
      size_t ro = (size_t)(so + a) * HDIM + n0 + wn + nl;
      unsigned short hi, lo;
      split_bf16(fmaxf(c0[v], 0.f), hi, lo);
      hbh[ro] = hi; hbl[ro] = lo;
      split_bf16(fmaxf(c1[v], 0.f), hi, lo);
      hbh[ro + 16] = hi; hbl[ro + 16] = lo;
    }
  }
}

// ---- kernel 6: grouped GEMM2  y[tok] += gate * (h @ w2[e]) -----------------
__global__ void moe_ffn2(const unsigned short* __restrict__ hbh,
                         const unsigned short* __restrict__ hbl,
                         const unsigned short* __restrict__ wh,
                         const unsigned short* __restrict__ wl,
                         const int* __restrict__ cnt,
                         const int* __restrict__ seg,
                         const int* __restrict__ idxbuf,
                         const float* __restrict__ gates,
                         float* __restrict__ y) {
  __shared__ __align__(16) unsigned short lAh[2][2048], lAl[2][2048];
  __shared__ __align__(16) unsigned short lBh[2][2048], lBl[2][2048];
  int e  = blockIdx.z;
  int mc = cnt[e];
  int m0 = blockIdx.x * 64;
  if (m0 >= mc) return;
  int n0 = blockIdx.y * 64;
  int so = seg[e];
  int t = threadIdx.x, lane = t & 31, wid = t >> 5;
  int wm = (wid & 3) * 16, wn = (wid >> 2) * 32;
  int r = t >> 2, c8 = (t & 3) * 8;
  const unsigned short* ah = hbh + (size_t)(so + m0 + r) * HDIM + c8;
  const unsigned short* al = hbl + (size_t)(so + m0 + r) * HDIM + c8;
  const unsigned short* bh = wh + (size_t)e * DDIM * HDIM + (size_t)n0 * HDIM;
  const unsigned short* bl = wl + (size_t)e * DDIM * HDIM + (size_t)n0 * HDIM;
  v8f c0 = {0.f, 0.f, 0.f, 0.f, 0.f, 0.f, 0.f, 0.f};
  v8f c1 = c0;

  if (wid == 0) {
    tdm_load_32x64(&lBh[0][0], bh);
    tdm_load_32x64(&lBl[0][0], bl);
  }
  async_cp16(&lAh[0][t * 8], ah);
  async_cp16(&lAl[0][t * 8], al);

  for (int it = 0; it < HDIM / 32; ++it) {
    int cur = it & 1, nxt = cur ^ 1;
    int k1 = (it + 1) * 32;
    if (k1 < HDIM) {
      if (wid == 0) {
        tdm_load_32x64(&lBh[nxt][0], bh + k1);
        tdm_load_32x64(&lBl[nxt][0], bl + k1);
      }
      async_cp16(&lAh[nxt][t * 8], ah + k1);
      async_cp16(&lAl[nxt][t * 8], al + k1);
      wait_async2();
      if (wid == 0) wait_tensor2();
    } else {
      wait_async0();
      if (wid == 0) wait_tensor0();
    }
    __syncthreads();
    v16bf fah = load_frag_a(&lAh[cur][0], lane, wm);
    v16bf fal = load_frag_a(&lAl[cur][0], lane, wm);
    v16bf bh0 = load_frag_b(&lBh[cur][0], lane, wn);
    v16bf bl0 = load_frag_b(&lBl[cur][0], lane, wn);
    v16bf bh1 = load_frag_b(&lBh[cur][0], lane, wn + 16);
    v16bf bl1 = load_frag_b(&lBl[cur][0], lane, wn + 16);
    c0 = wmma_split(c0, fah, fal, bh0, bl0);
    c1 = wmma_split(c1, fah, fal, bh1, bl1);
    __syncthreads();
  }

  // each y element receives exactly 2 commutative f32 atomic adds onto zero
  int ml = (lane >> 4) * 8, nl = lane & 15;
#pragma unroll
  for (int v = 0; v < 8; ++v) {
    int a = m0 + wm + ml + v;
    if (a < mc) {
      int tok = idxbuf[e * N_TOK + a];
      float g = gates[tok * NEXP + e];
      int col = n0 + wn + nl;
      atomicAdd(&y[(size_t)tok * DDIM + col],      g * c0[v]);
      atomicAdd(&y[(size_t)tok * DDIM + col + 16], g * c1[v]);
    }
  }
}

// ---- kernel 7: aux losses (fixed-order deterministic reductions) -----------
__global__ void moe_loss(const float* __restrict__ probs,
                         const float* __restrict__ gates,
                         const float* __restrict__ lse2,
                         float* __restrict__ out_loss) {
  __shared__ float red[256];
  int t = threadIdx.x;
  float part[25];
#pragma unroll
  for (int q = 0; q < 25; ++q) part[q] = 0.f;
  for (int n = t; n < N_TOK; n += 256) {
#pragma unroll
    for (int e = 0; e < NEXP; ++e) {
      float pv = probs[n * NEXP + e];
      float gv = gates[n * NEXP + e];
      part[e]      += pv;
      part[8 + e]  += gv;
      part[16 + e] += (gv > 0.f) ? 1.f : 0.f;
    }
    part[24] += lse2[n];
  }
  float tot[25];
  for (int q = 0; q < 25; ++q) {
    red[t] = part[q];
    __syncthreads();
    for (int s = 128; s > 0; s >>= 1) {
      if (t < s) red[t] += red[t + s];
      __syncthreads();
    }
    tot[q] = red[0];
    __syncthreads();
  }
  if (t == 0) {
    float pt = 0.f, gt = 0.f, et = 0.f;
    for (int e = 0; e < NEXP; ++e) { pt += tot[e]; gt += tot[8 + e]; et += tot[16 + e]; }
    pt = fmaxf(pt, 1e-12f); gt = fmaxf(gt, 1e-12f); et = fmaxf(et, 1e-12f);
    float gn[NEXP], mean = 0.f;
    for (int e = 0; e < NEXP; ++e) { gn[e] = tot[8 + e] / gt; mean += gn[e]; }
    mean /= (float)NEXP;
    float var = 0.f;
    for (int e = 0; e < NEXP; ++e) { float d = gn[e] - mean; var += d * d; }
    var /= (float)(NEXP - 1);
    float cv = var / (mean * mean + 1e-10f);
    float sw = 0.f;
    for (int e = 0; e < NEXP; ++e) sw += (tot[e] / pt) * (tot[16 + e] / et);
    sw = (1.f - sw) * (float)NEXP;
    float z = tot[24] / (float)N_TOK;
    out_loss[0] = 0.01f * cv + 0.1f * sw + 1e-4f * z;
  }
}

// ---------------------------------------------------------------------------
#define OFF_CNT    0
#define OFF_SEG    64
#define OFF_IDX    256
#define OFF_GATES  131584
#define OFF_PROBS  262912
#define OFF_LSE2   394240
#define OFF_XS_HI  417792ull
#define OFF_XS_LO  (OFF_XS_HI + 8388608ull)
#define OFF_W1T_HI (OFF_XS_LO + 8388608ull)
#define OFF_W1T_LO (OFF_W1T_HI + 16777216ull)
#define OFF_W2T_HI (OFF_W1T_LO + 16777216ull)
#define OFF_W2T_LO (OFF_W2T_HI + 16777216ull)
#define OFF_HB_HI  (OFF_W2T_LO + 16777216ull)
#define OFF_HB_LO  (OFF_HB_HI + (unsigned long long)HPAD_ROWS * HDIM * 2ull)

extern "C" void kernel_launch(void* const* d_in, const int* in_sizes, int n_in,
                              void* d_out, int out_size, void* d_ws,
                              size_t ws_size, hipStream_t stream) {
  const float* x   = (const float*)d_in[0];
  const float* wg  = (const float*)d_in[1];
  const float* w1  = (const float*)d_in[2];
  const float* w2  = (const float*)d_in[3];
  const float* eps = (const float*)d_in[4];

  float* y    = (float*)d_out;
  float* loss = y + (size_t)N_TOK * DDIM;

  char* ws = (char*)d_ws;
  int*            cnt    = (int*)(ws + OFF_CNT);
  int*            seg    = (int*)(ws + OFF_SEG);
  int*            idxbuf = (int*)(ws + OFF_IDX);
  float*          gates  = (float*)(ws + OFF_GATES);
  float*          probs  = (float*)(ws + OFF_PROBS);
  float*          lse2   = (float*)(ws + OFF_LSE2);
  unsigned short* xs_hi  = (unsigned short*)(ws + OFF_XS_HI);
  unsigned short* xs_lo  = (unsigned short*)(ws + OFF_XS_LO);
  unsigned short* w1t_hi = (unsigned short*)(ws + OFF_W1T_HI);
  unsigned short* w1t_lo = (unsigned short*)(ws + OFF_W1T_LO);
  unsigned short* w2t_hi = (unsigned short*)(ws + OFF_W2T_HI);
  unsigned short* w2t_lo = (unsigned short*)(ws + OFF_W2T_LO);
  unsigned short* hb_hi  = (unsigned short*)(ws + OFF_HB_HI);
  unsigned short* hb_lo  = (unsigned short*)(ws + OFF_HB_LO);

  moe_zero_y<<<dim3(4096), dim3(256), 0, stream>>>((float4*)y);
  moe_prep_x<<<dim3(4096), dim3(256), 0, stream>>>((const float4*)x, xs_hi, xs_lo);
  moe_prep_w<<<dim3(16, 16, 16), dim3(256), 0, stream>>>(w1, w2, w1t_hi, w1t_lo,
                                                         w2t_hi, w2t_lo);
  moe_gating<<<dim3(N_TOK / 8), dim3(256), 0, stream>>>(x, wg, eps, gates,
                                                        probs, lse2);
  moe_compact<<<dim3(1), dim3(256), 0, stream>>>(gates, cnt, seg, idxbuf);
  moe_ffn1<<<dim3(64, HDIM / 64, NEXP), dim3(256), 0, stream>>>(
      xs_hi, xs_lo, w1t_hi, w1t_lo, cnt, seg, idxbuf, hb_hi, hb_lo);
  moe_ffn2<<<dim3(64, DDIM / 64, NEXP), dim3(256), 0, stream>>>(
      hb_hi, hb_lo, w2t_hi, w2t_lo, cnt, seg, idxbuf, gates, y);
  moe_loss<<<dim3(1), dim3(256), 0, stream>>>(probs, gates, lse2, loss);
}
